// GraphAttention_51762945852111
// MI455X (gfx1250) — matrix operand order repaired and checked by reading, not compiled, not run on hardware
//
#include <hip/hip_runtime.h>

// ---------------------------------------------------------------------------
// Graph attention: out = softmax(where(A>0, X X^T, -1e9)) X
// N=12288, F=512, fp32 in/out.  bf16 WMMA flash-attention for MI455X/gfx1250.
// Double-buffered LDS with async global->LDS staging, double-buffered WMMA
// B-operands, wave-private online softmax.
// ---------------------------------------------------------------------------

#define N_ 12288
#define F_ 512

typedef __attribute__((ext_vector_type(16))) __bf16 v16bf;
typedef __attribute__((ext_vector_type(8)))  __bf16 v8bf;
typedef __attribute__((ext_vector_type(8)))  float  v8f;

// pointer types for the async global->LDS builtin (signature revealed by
// the round-2 diagnostic: param0 is AS1 pointer to int __vector_size__(16))
typedef int v4i_ __attribute__((vector_size(16)));
typedef __attribute__((address_space(1))) v4i_* gv4i_p;
typedef __attribute__((address_space(3))) v4i_* lv4i_p;

constexpr int BM = 64;            // query rows per workgroup (4 waves x 16)
constexpr int BN = 64;            // keys per iteration
constexpr int KB_PITCH = F_ + 8;  // 520 bf16 -> 1040B row pitch (conflict-free)
constexpr int VT_PITCH = BN + 8;  // 72  bf16 -> 144B row pitch
constexpr int P_PITCH  = BN + 8;  // 72

// [Kb0][Kb1][VT0][VT1][P(4 waves)]
constexpr size_t KB_ELEMS = (size_t)BN * KB_PITCH;   // 33280
constexpr size_t VT_ELEMS = (size_t)F_ * VT_PITCH;   // 36864
constexpr size_t P_ELEMS  = 4ull * 16 * P_PITCH;     // 4608
constexpr size_t SMEM_ELEMS = 2 * KB_ELEMS + 2 * VT_ELEMS + P_ELEMS;
constexpr size_t SMEM_BYTES = SMEM_ELEMS * sizeof(__bf16);   // 289,792 B

#if defined(__has_builtin)
#if __has_builtin(__builtin_amdgcn_global_load_async_to_lds_b128) && \
    __has_builtin(__builtin_amdgcn_s_wait_asynccnt)
#define USE_ASYNC_LDS 1
#endif
#endif
#ifndef USE_ASYNC_LDS
#define USE_ASYNC_LDS 0
#endif

__device__ __forceinline__ v16bf pack16(v8bf lo, v8bf hi) {
    union { v16bf v; v8bf h[2]; } u;
    u.h[0] = lo; u.h[1] = hi;
    return u.v;
}

// ---------------------------------------------------------------------------
// Kernel 1: fp32 features -> bf16 row-major copy Xb and bf16 transpose XbT.
// ---------------------------------------------------------------------------
__global__ __launch_bounds__(256) void convert_kernel(
        const float* __restrict__ X, __bf16* __restrict__ Xb,
        __bf16* __restrict__ XbT) {
    __shared__ float tile[32][33];
    const int tx = threadIdx.x, ty = threadIdx.y;      // 32 x 8
    const int f0 = blockIdx.x * 32, n0 = blockIdx.y * 32;
#pragma unroll
    for (int i = 0; i < 4; ++i) {
        float v = X[(size_t)(n0 + ty + 8 * i) * F_ + f0 + tx];
        tile[ty + 8 * i][tx] = v;
        Xb[(size_t)(n0 + ty + 8 * i) * F_ + f0 + tx] = (__bf16)v;
    }
    __syncthreads();
#pragma unroll
    for (int i = 0; i < 4; ++i)
        XbT[(size_t)(f0 + ty + 8 * i) * N_ + n0 + tx] =
            (__bf16)tile[tx][ty + 8 * i];
}

// ---------------------------------------------------------------------------
// Kernel 2: flash attention.
// ---------------------------------------------------------------------------
__global__ __launch_bounds__(128) void attn_kernel(
        const __bf16* __restrict__ Xb, const __bf16* __restrict__ XbT,
        const int* __restrict__ Adj, float* __restrict__ Out) {
    extern __shared__ __align__(16) char smem_raw[];
    __bf16* const KbBuf0 = (__bf16*)smem_raw;
    __bf16* const KbBuf1 = KbBuf0 + KB_ELEMS;
    __bf16* const VTBuf0 = KbBuf1 + KB_ELEMS;
    __bf16* const VTBuf1 = VTBuf0 + VT_ELEMS;
    __bf16* const Pb     = VTBuf1 + VT_ELEMS;   // [4][16][P_PITCH]

    const int tid  = threadIdx.x;
    const int wv   = tid >> 5;
    const int lane = tid & 31;
    const int l16  = lane & 15;
    const int hi   = lane >> 4;
    const int qRow = blockIdx.x * BM + wv * 16;

    __bf16* const Pw = Pb + (size_t)wv * 16 * P_PITCH;

    // ---- async (or fallback) staging of one key/value block --------------
    auto stage_block = [&](int kb, __bf16* KbD, __bf16* VTD) {
#if USE_ASYNC_LDS
#pragma unroll 4
        for (int i = tid; i < BN * (F_ / 8); i += 128) {
            const int row = i >> 6, ch = i & 63;
            __builtin_amdgcn_global_load_async_to_lds_b128(
                (gv4i_p)(void*)(Xb + (size_t)(kb + row) * F_ + ch * 8),
                (lv4i_p)(void*)(KbD + (size_t)row * KB_PITCH + ch * 8),
                0, 0);
        }
#pragma unroll 4
        for (int i = tid; i < F_ * (BN / 8); i += 128) {
            const int row = i >> 3, ch = i & 7;
            __builtin_amdgcn_global_load_async_to_lds_b128(
                (gv4i_p)(void*)(XbT + (size_t)row * N_ + kb + ch * 8),
                (lv4i_p)(void*)(VTD + (size_t)row * VT_PITCH + ch * 8),
                0, 0);
        }
#else
#pragma unroll 4
        for (int i = tid; i < BN * (F_ / 8); i += 128) {
            const int row = i >> 6, ch = i & 63;
            *(v8bf*)(KbD + (size_t)row * KB_PITCH + ch * 8) =
                *(const v8bf*)(Xb + (size_t)(kb + row) * F_ + ch * 8);
        }
#pragma unroll 4
        for (int i = tid; i < F_ * (BN / 8); i += 128) {
            const int row = i >> 3, ch = i & 7;
            *(v8bf*)(VTD + (size_t)row * VT_PITCH + ch * 8) =
                *(const v8bf*)(XbT + (size_t)row * N_ + kb + ch * 8);
        }
#endif
    };
    auto buffer_sync = [&]() {
#if USE_ASYNC_LDS
        __builtin_amdgcn_s_wait_asynccnt(0);
#endif
        __syncthreads();
    };

    // ---- persistent state -------------------------------------------------
    const v8f vzero = {0.f, 0.f, 0.f, 0.f, 0.f, 0.f, 0.f, 0.f};
    v8f acc[32];
#pragma unroll
    for (int i = 0; i < 32; ++i) acc[i] = vzero;

    float mrow[8], lrow[8];
#pragma unroll
    for (int r = 0; r < 8; ++r) { mrow[r] = -INFINITY; lrow[r] = 0.f; }

    // Q resident in 128 VGPRs for the whole kernel (loop-invariant A operand)
    const __bf16* const qbase = Xb + (size_t)(qRow + l16) * F_;
    v16bf qreg[16];
#pragma unroll
    for (int fc = 0; fc < F_; fc += 32)
        qreg[fc / 32] = pack16(*(const v8bf*)(qbase + fc + 8 * hi),
                               *(const v8bf*)(qbase + fc + 16 + 8 * hi));

    __bf16 *KbC = KbBuf0, *VTC = VTBuf0, *KbN = KbBuf1, *VTN = VTBuf1;
    stage_block(0, KbC, VTC);

    for (int kb = 0; kb < N_; kb += BN) {
        buffer_sync();                 // buf(cur) staged; buf(next) free
        if (kb + BN < N_) stage_block(kb + BN, KbN, VTN);

        // prefetch the adjacency ints for this tile (only HBM stream)
        int av[4][8];
#pragma unroll
        for (int r = 0; r < 8; ++r) {
            const int* arow = Adj + (size_t)(qRow + r + 8 * hi) * N_ + kb + l16;
#pragma unroll
            for (int t = 0; t < 4; ++t)
                av[t][r] = __builtin_nontemporal_load(arow + t * 16);
        }

        // ---- scores: S[t] (16x16) = Q . K^T, double-buffered B ----------
        v8f S[4];
#pragma unroll
        for (int t = 0; t < 4; ++t) S[t] = vzero;

        auto loadKB = [&](int t, int fc) -> v16bf {
            const __bf16* bb =
                KbC + (size_t)(t * 16 + l16) * KB_PITCH + fc + 8 * hi;
            return pack16(*(const v8bf*)bb, *(const v8bf*)(bb + 16));
        };
        v16bf bc[4], bn[4];
#pragma unroll
        for (int t = 0; t < 4; ++t) bc[t] = loadKB(t, 0);
#pragma unroll
        for (int fc = 0; fc < F_; fc += 32) {
            if (fc + 32 < F_) {
#pragma unroll
                for (int t = 0; t < 4; ++t) bn[t] = loadKB(t, fc + 32);
            }
            const v16bf a = qreg[fc / 32];
#pragma unroll
            for (int t = 0; t < 4; ++t)
                S[t] = __builtin_amdgcn_wmma_f32_16x16x32_bf16(
                    false, a, false, bc[t], (short)0, S[t], false, false);
            if (fc + 32 < F_) {
#pragma unroll
                for (int t = 0; t < 4; ++t) bc[t] = bn[t];
            }
        }

        // ---- mask + wave-private online softmax --------------------------
        float mloc[8];
#pragma unroll
        for (int r = 0; r < 8; ++r) {
            float mx = -INFINITY;
#pragma unroll
            for (int t = 0; t < 4; ++t) {
                float s = S[t][r];
                s = (av[t][r] > 0) ? s : -1.0e9f;
                S[t][r] = s;
                mx = fmaxf(mx, s);
            }
            mloc[r] = mx;
        }
#pragma unroll
        for (int m = 1; m < 16; m <<= 1)
#pragma unroll
            for (int r = 0; r < 8; ++r)
                mloc[r] = fmaxf(mloc[r], __shfl_xor(mloc[r], m, 32));

        float scale[8];
#pragma unroll
        for (int r = 0; r < 8; ++r) {
            const float mnew = fmaxf(mrow[r], mloc[r]);
            scale[r] = __expf(mrow[r] - mnew);   // exp(-inf)=0 on first block
            mrow[r] = mnew;
        }

        float ps[8];
#pragma unroll
        for (int r = 0; r < 8; ++r) {
            float sum = 0.f;
#pragma unroll
            for (int t = 0; t < 4; ++t) {
                const float p = __expf(S[t][r] - mrow[r]);
                sum += p;
                Pw[(size_t)(r + 8 * hi) * P_PITCH + t * 16 + l16] = (__bf16)p;
            }
            ps[r] = sum;
        }
#pragma unroll
        for (int m = 1; m < 16; m <<= 1)
#pragma unroll
            for (int r = 0; r < 8; ++r) ps[r] += __shfl_xor(ps[r], m, 32);
#pragma unroll
        for (int r = 0; r < 8; ++r) lrow[r] = lrow[r] * scale[r] + ps[r];

#pragma unroll
        for (int ft = 0; ft < 32; ++ft)
#pragma unroll
            for (int r = 0; r < 8; ++r) acc[ft][r] *= scale[r];

        // ---- P.V : acc(16x512) += P(16x64) . V(64x512) -------------------
        // (same-wave DS ops are in-order; P slab is wave-private)
        auto loadVT = [&](int ft, int c) -> v16bf {
            const __bf16* vb =
                VTC + (size_t)(ft * 16 + l16) * VT_PITCH + c * 32 + 8 * hi;
            return pack16(*(const v8bf*)vb, *(const v8bf*)(vb + 16));
        };
#pragma unroll
        for (int c = 0; c < 2; ++c) {
            const __bf16* pl = Pw + (size_t)l16 * P_PITCH + c * 32;
            const v16bf pa = pack16(*(const v8bf*)(pl + 8 * hi),
                                    *(const v8bf*)(pl + 16 + 8 * hi));
            v16bf vc[4], vn[4];
#pragma unroll
            for (int j = 0; j < 4; ++j) vc[j] = loadVT(j, c);
#pragma unroll
            for (int g = 0; g < 8; ++g) {
                if (g < 7) {
#pragma unroll
                    for (int j = 0; j < 4; ++j)
                        vn[j] = loadVT(4 * (g + 1) + j, c);
                }
#pragma unroll
                for (int j = 0; j < 4; ++j)
                    acc[4 * g + j] = __builtin_amdgcn_wmma_f32_16x16x32_bf16(
                        false, pa, false, vc[j], (short)0, acc[4 * g + j],
                        false, false);
                if (g < 7) {
#pragma unroll
                    for (int j = 0; j < 4; ++j) vc[j] = vn[j];
                }
            }
        }

        // swap LDS double buffers
        __bf16* tk = KbC; KbC = KbN; KbN = tk;
        __bf16* tv = VTC; VTC = VTN; VTN = tv;
    }

    // ---- epilogue: out = acc / l, non-temporal stores --------------------
#pragma unroll
    for (int r = 0; r < 8; ++r) {
        const float rcp = 1.0f / lrow[r];
        const int grow = qRow + r + 8 * hi;
#pragma unroll
        for (int ft = 0; ft < 32; ++ft)
            __builtin_nontemporal_store(
                acc[ft][r] * rcp, Out + (size_t)grow * F_ + ft * 16 + l16);
    }
}

// ---------------------------------------------------------------------------
extern "C" void kernel_launch(void* const* d_in, const int* in_sizes, int n_in,
                              void* d_out, int out_size, void* d_ws,
                              size_t ws_size, hipStream_t stream) {
    const float* features = (const float*)d_in[0];
    const int*   A        = (const int*)d_in[1];
    float*       out      = (float*)d_out;

    __bf16* Xb  = (__bf16*)d_ws;                 // N x F  bf16  (12.6 MB)
    __bf16* XbT = Xb + (size_t)N_ * F_;          // F x N  bf16  (12.6 MB)

    dim3 cb(32, 8);
    dim3 cg(F_ / 32, N_ / 32);
    convert_kernel<<<cg, cb, 0, stream>>>(features, Xb, XbT);

    dim3 ab(128);
    dim3 ag(N_ / BM);   // 192 workgroups
    attn_kernel<<<ag, ab, SMEM_BYTES, stream>>>(Xb, XbT, A, out);
}
